// UpsamplingBlock_10565619548540
// MI455X (gfx1250) — matrix operand order repaired and checked by reading, not compiled, not run on hardware
//
#include <hip/hip_runtime.h>
#include <hip/hip_bf16.h>
#include <stdint.h>

typedef __attribute__((ext_vector_type(16))) _Float16 v16h;
typedef __attribute__((ext_vector_type(8)))  _Float16 v8h;
typedef __attribute__((ext_vector_type(8)))  float    v8f;

#define HW_   16384   // 128*128
#define W_    128
#define H_    128

__device__ __forceinline__ float signf_(float v) {
    return (v > 0.f) ? 1.f : ((v < 0.f) ? -1.f : 0.f);
}

// ---------------------------------------------------------------------------
// Bilinear sample from a 64x64 plane at output coords (y,x) of a 128x128 grid
// (half-pixel centers: src = out*0.5 - 0.25, clamped).
// ---------------------------------------------------------------------------
__device__ __forceinline__ float upsample64_(const float* __restrict__ base, int y, int x) {
    float sy = y * 0.5f - 0.25f;
    float sx = x * 0.5f - 0.25f;
    int   y0 = (int)floorf(sy);
    int   x0 = (int)floorf(sx);
    float wy = sy - (float)y0;
    float wx = sx - (float)x0;
    int y0c = min(max(y0, 0), 63), y1c = min(max(y0 + 1, 0), 63);
    int x0c = min(max(x0, 0), 63), x1c = min(max(x0 + 1, 0), 63);
    float v00 = base[y0c * 64 + x0c], v01 = base[y0c * 64 + x1c];
    float v10 = base[y1c * 64 + x0c], v11 = base[y1c * 64 + x1c];
    return v00 * (1.f - wy) * (1.f - wx) + v01 * (1.f - wy) * wx +
           v10 * wy * (1.f - wx)        + v11 * wy * wx;
}

// ---------------------------------------------------------------------------
// Build sign-activation for the main branch convs:
//   act[p][c] = sign( xc[p][c] + lb1[c] ),  NHWC f16, CCAT = 256
//   xc = concat( bilinear_up(x), f, s )
// ---------------------------------------------------------------------------
__global__ void prep_act_main(const float* __restrict__ xin,  // (8,128,64,64)
                              const float* __restrict__ f,    // (8,64,128,128)
                              const float* __restrict__ s,    // (8,64,128,128)
                              const float* __restrict__ lb1,  // (256)
                              _Float16* __restrict__ act) {
    int p   = blockIdx.x;         // 0..131071
    int c   = threadIdx.x;        // 0..255
    int img = p >> 14;
    int y   = (p >> 7) & 127;
    int x   = p & 127;
    float v;
    if (c < 128) {
        const float* base = xin + ((size_t)img * 128 + c) * 4096;
        v = upsample64_(base, y, x);
    } else if (c < 192) {
        v = f[((size_t)img * 64 + (c - 128)) * HW_ + (y << 7) + x];
    } else {
        v = s[((size_t)img * 64 + (c - 192)) * HW_ + (y << 7) + x];
    }
    v += lb1[c];
    act[(size_t)p * 256 + c] = (_Float16)signf_(v);
}

// Bilinear-resize inc_mask (8,1,64,64) -> msk (8,16384)
__global__ void prep_msk_kernel(const float* __restrict__ inc, float* __restrict__ msk) {
    int idx = blockIdx.x * 256 + threadIdx.x;
    if (idx >= 8 * HW_) return;
    int img = idx >> 14;
    int y   = (idx >> 7) & 127;
    int x   = idx & 127;
    msk[idx] = upsample64_(inc + (size_t)img * 4096, y, x);
}

// ---------------------------------------------------------------------------
// Binarize weights: raw OIHW float -> wp[o][k] = sign(w - mean_o) (f16),
// k = tap*cin + i.  sf[o] = mean(|w - mean_o|).  One block per out channel.
// ---------------------------------------------------------------------------
__global__ void pack_weights(const float* __restrict__ w, _Float16* __restrict__ wp,
                             float* __restrict__ sf, int cin, int khw) {
    int o = blockIdx.x;
    int K = cin * khw;
    __shared__ float red[256];
    __shared__ float meanv;
    const float* wo = w + (size_t)o * K;

    float acc = 0.f;
    for (int e = threadIdx.x; e < K; e += 256) acc += wo[e];
    red[threadIdx.x] = acc;
    __syncthreads();
    for (int st = 128; st > 0; st >>= 1) {
        if (threadIdx.x < st) red[threadIdx.x] += red[threadIdx.x + st];
        __syncthreads();
    }
    if (threadIdx.x == 0) meanv = red[0] / (float)K;
    __syncthreads();
    float mu = meanv;

    acc = 0.f;
    for (int e = threadIdx.x; e < K; e += 256) acc += fabsf(wo[e] - mu);
    red[threadIdx.x] = acc;
    __syncthreads();
    for (int st = 128; st > 0; st >>= 1) {
        if (threadIdx.x < st) red[threadIdx.x] += red[threadIdx.x + st];
        __syncthreads();
    }
    if (threadIdx.x == 0) sf[o] = red[0] / (float)K;

    for (int kidx = threadIdx.x; kidx < K; kidx += 256) {
        int tap = kidx / cin;
        int i   = kidx - tap * cin;
        float wv = wo[(size_t)i * khw + tap] - mu;
        wp[(size_t)o * K + kidx] = (_Float16)signf_(wv);
    }
}

// Fold sf + BN + lb2 into per-channel affine: y = acc*alpha + beta, then prelu
__global__ void make_branch_affine(const float* sf, const float* g, const float* b,
                                   const float* m, const float* vv, const float* lb2,
                                   float* alpha, float* beta, int n) {
    int c = threadIdx.x;
    if (c >= n) return;
    float sc = g[c] * rsqrtf(vv[c] + 1e-5f);
    alpha[c] = sf[c] * sc;
    beta[c]  = b[c] - m[c] * sc + lb2[c];
}

// Gate affine: y = acc*sf + bias + lb2
__global__ void make_gate_affine(const float* sf, const float* b, const float* lb2,
                                 float* alpha, float* beta, int n) {
    int c = threadIdx.x;
    if (c >= n) return;
    alpha[c] = sf[c];
    beta[c]  = b[c] + lb2[c];
}

// ---------------------------------------------------------------------------
// Implicit-GEMM conv via V_WMMA_F32_16X16X32_F16.
// One wave -> 16 pixels x (NT*16) output channels: the A fragment (bounds-
// checked activation gather) is built once per K-step and reused across NT
// WMMAs, so the loop issues NT wmma per (2 A-loads + NT B-loads).
// act  : [npix][CIN]  NHWC f16 (+-1 signs)
// wgt  : [COUTC][K]   f16 (+-1 signs), K = KH*KW*CIN, tap-major channel-minor
// out  : NCHW f32 (img,COUTC,128,128), per-mode epilogue.
// A fragment (16-bit 16x32, ISA 7.12.2): lane L holds row M=L%16; K-runs
// [khalf*16,+8) and [khalf*16+8,+8) (contiguous k, same as B), khalf=L/16 -> two aligned v8h loads.
// B fragment: lane L holds col N=L%16; K-run [khalf*16,+16) -> one v16h load.
// Modes: 0 prelu*mask store | 1 prelu accumulate | 2 sigmoid store
//        3 prev*mask+sigmoid | 4 tanh store | 5 prev*mask+tanh
// ---------------------------------------------------------------------------
template <int CIN, int KH, int KW, int PAD, int NT>
__global__ __launch_bounds__(128) void conv_wmma(
    const _Float16* __restrict__ act, const _Float16* __restrict__ wgt,
    const float* __restrict__ alpha, const float* __restrict__ beta,
    const float* __restrict__ prelu_a, const float* __restrict__ mask,
    const float* __restrict__ prev, float* __restrict__ out,
    int npix, int mode) {
    constexpr int K     = KH * KW * CIN;
    constexpr int COUTC = NT * 16;
    const int lane   = threadIdx.x & 31;      // wave32
    const int wave   = threadIdx.x >> 5;
    const int gw     = blockIdx.x * 4 + wave; // one wave = one 16-pixel tile
    const int mtiles = npix >> 4;
    if (gw >= mtiles) return;                 // wave-uniform: EXEC stays all-1

    const int m     = lane & 15;
    const int khalf = lane >> 4;
    const int p0    = gw << 4;
    const int pm    = p0 + m;                 // this lane's pixel row of A
    const int img   = pm >> 14;
    const int y     = (pm >> 7) & 127;
    const int x     = pm & 127;
    const int n0    = lane & 15;              // B column / D channel base
    const _Float16* wbase = wgt + (size_t)n0 * K + (khalf << 4);

    v8f acc[NT] = {};
    for (int k0 = 0; k0 < K; k0 += 32) {
        union { v16h v; v8h h[2]; } A;
#pragma unroll
        for (int half = 0; half < 2; ++half) {
            int kb  = k0 + (khalf << 4) + half * 8;
            int tap = kb / CIN;
            int c   = kb - tap * CIN;
            int dy  = tap / KW - PAD;
            int dx  = tap - (tap / KW) * KW - PAD;
            int yy = y + dy, xx = x + dx;
            v8h av = {};
            if (KH == 1 || ((unsigned)yy < (unsigned)H_ && (unsigned)xx < (unsigned)W_)) {
                size_t sidx = ((size_t)(img << 14) + (yy << 7) + xx) * CIN + c;
                av = *(const v8h*)(act + sidx);
            }
            A.h[half] = av;
        }
        if (k0 + 256 < K)  // stream weights ahead (global_prefetch_b8)
            __builtin_prefetch((const void*)(wbase + k0 + 256), 0, 1);
#pragma unroll
        for (int j = 0; j < NT; ++j) {
            v16h B = *(const v16h*)(wbase + (size_t)j * 16 * K + k0);
            acc[j] = __builtin_amdgcn_wmma_f32_16x16x32_f16(
                false, A.v, false, B, (short)0, acc[j], false, false);
        }
    }

    // Epilogue. D layout: lane group g=lane/16, VGPR v -> M = v + 8*g, N = lane%16.
    const int rowoff = khalf << 3;
#pragma unroll
    for (int j = 0; j < NT; ++j) {
        int   nch = j * 16 + n0;
        float al  = alpha[nch];
        float be  = beta[nch];
        float pa  = (mode <= 1) ? prelu_a[nch] : 0.f;
#pragma unroll
        for (int v = 0; v < 8; ++v) {
            int p  = p0 + rowoff + v;
            int im = p >> 14;
            int pp = p & (HW_ - 1);
            size_t oidx = (size_t)im * COUTC * HW_ + (size_t)nch * HW_ + pp;
            float yv = acc[j][v] * al + be;
            float res;
            if (mode == 0) {
                res = (yv >= 0.f ? yv : pa * yv) * mask[p];
            } else if (mode == 1) {
                res = out[oidx] + (yv >= 0.f ? yv : pa * yv);
            } else if (mode == 2) {
                res = 1.f / (1.f + __expf(-yv));
            } else if (mode == 3) {
                res = prev[oidx] * mask[p] + 1.f / (1.f + __expf(-yv));
            } else if (mode == 4) {
                res = tanhf(yv);
            } else {
                res = prev[oidx] * mask[p] + tanhf(yv);
            }
            out[oidx] = res;
        }
    }
}

// ---------------------------------------------------------------------------
// GRU helper kernels
// ---------------------------------------------------------------------------
// xh = concat(b_t, h); act = sign(xh + lb1), NHWC f16 (16384 x 64)
__global__ void prep_xh_act(const float* __restrict__ xn_t,   // (64,16384), b = ch 32..63
                            const float* __restrict__ h,      // (32,16384)
                            const float* __restrict__ lb1,    // (64)
                            _Float16* __restrict__ act) {
    int p = blockIdx.x;
    int c = threadIdx.x;  // 0..63
    float v = (c < 32) ? xn_t[(size_t)(32 + c) * HW_ + p]
                       : h[(size_t)(c - 32) * HW_ + p];
    act[(size_t)p * 64 + c] = (_Float16)signf_(v + lb1[c]);
}

// xr = concat(b_t, r*h); r = rz channels 0..31
__global__ void prep_xr_act(const float* __restrict__ xn_t,
                            const float* __restrict__ h,
                            const float* __restrict__ rz,
                            const float* __restrict__ lb1,
                            _Float16* __restrict__ act) {
    int p = blockIdx.x;
    int c = threadIdx.x;
    float v;
    if (c < 32) {
        v = xn_t[(size_t)(32 + c) * HW_ + p];
    } else {
        int cc = c - 32;
        v = rz[(size_t)cc * HW_ + p] * h[(size_t)cc * HW_ + p];
    }
    act[(size_t)p * 64 + c] = (_Float16)signf_(v + lb1[c]);
}

// h = (1-z)*h + z*c ; also write timestep output channels 32..63
__global__ void gru_update(float* __restrict__ h, const float* __restrict__ rz,
                           const float* __restrict__ cb, float* __restrict__ out_t) {
    int idx = blockIdx.x * 256 + threadIdx.x;
    if (idx >= 32 * HW_) return;
    int ch = idx >> 14;
    int p  = idx & (HW_ - 1);
    float z  = rz[(size_t)(32 + ch) * HW_ + p];
    float cv = cb[idx];
    float hn = (1.f - z) * h[idx] + z * cv;
    h[idx] = hn;
    out_t[(size_t)(32 + ch) * HW_ + p] = hn;
}

__global__ void copyf_kernel(const float* __restrict__ src, float* __restrict__ dst, int n) {
    int idx = blockIdx.x * 256 + threadIdx.x;
    if (idx < n) dst[idx] = src[idx];
}

// Copy 'a' half (channels 0..31 of each frame) from xn to d_out
__global__ void copy_a_kernel(const float* __restrict__ xn, float* __restrict__ out) {
    int idx = blockIdx.x * 256 + threadIdx.x;
    if (idx >= 8 * 32 * HW_) return;
    int t   = idx / (32 * HW_);
    int rem = idx - t * (32 * HW_);
    int ch  = rem >> 14;
    int p   = rem & (HW_ - 1);
    size_t k = ((size_t)(t * 64 + ch) << 14) + p;
    out[k] = xn[k];
}

// ---------------------------------------------------------------------------
extern "C" void kernel_launch(void* const* d_in, const int* in_sizes, int n_in,
                              void* d_out, int out_size, void* d_ws, size_t ws_size,
                              hipStream_t stream) {
    (void)in_sizes; (void)n_in; (void)out_size; (void)ws_size;
    // Inputs (setup_inputs order, params flattened depth-first in insertion order)
    const float* x    = (const float*)d_in[0];   // (1,8,128,64,64)
    const float* f    = (const float*)d_in[1];   // (1,8,64,128,128)
    const float* s    = (const float*)d_in[2];   // (1,8,64,128,128)
    const float* r0   = (const float*)d_in[3];   // (1,32,128,128)
    const float* incm = (const float*)d_in[4];   // (8,1,64,64)
    // params['conv'] (3x3 branch): lb1,w,bn_g,bn_b,bn_m,bn_v,lb2,prelu
    const float* c3_lb1 = (const float*)d_in[5];
    const float* c3_w   = (const float*)d_in[6];
    const float* c3_bng = (const float*)d_in[7];
    const float* c3_bnb = (const float*)d_in[8];
    const float* c3_bnm = (const float*)d_in[9];
    const float* c3_bnv = (const float*)d_in[10];
    const float* c3_lb2 = (const float*)d_in[11];
    const float* c3_pr  = (const float*)d_in[12];
    // params['conv1x1']
    const float* c1_lb1 = (const float*)d_in[13];
    const float* c1_w   = (const float*)d_in[14];
    const float* c1_bng = (const float*)d_in[15];
    const float* c1_bnb = (const float*)d_in[16];
    const float* c1_bnm = (const float*)d_in[17];
    const float* c1_bnv = (const float*)d_in[18];
    const float* c1_lb2 = (const float*)d_in[19];
    const float* c1_pr  = (const float*)d_in[20];
    // params['gru']: ih, ih1, hh, hh1 each {lb1,w,b,lb2}
    const float* ih_lb1  = (const float*)d_in[21];
    const float* ih_w    = (const float*)d_in[22];
    const float* ih_b    = (const float*)d_in[23];
    const float* ih_lb2  = (const float*)d_in[24];
    const float* ih1_lb1 = (const float*)d_in[25];
    const float* ih1_w   = (const float*)d_in[26];
    const float* ih1_b   = (const float*)d_in[27];
    const float* ih1_lb2 = (const float*)d_in[28];
    const float* hh_lb1  = (const float*)d_in[29];
    const float* hh_w    = (const float*)d_in[30];
    const float* hh_b    = (const float*)d_in[31];
    const float* hh_lb2  = (const float*)d_in[32];
    const float* hh1_lb1 = (const float*)d_in[33];
    const float* hh1_w   = (const float*)d_in[34];
    const float* hh1_b   = (const float*)d_in[35];
    const float* hh1_lb2 = (const float*)d_in[36];

    float* out = (float*)d_out;

    // Workspace carving (256B aligned)
    uint8_t* wsb = (uint8_t*)d_ws;
    size_t off = 0;
    auto alloc = [&](size_t n) -> void* {
        void* p = wsb + off;
        off = (off + n + 255) & ~(size_t)255;
        return p;
    };
    _Float16* act   = (_Float16*)alloc((size_t)131072 * 256 * 2);  // 64 MB
    float*    xn    = (float*)alloc((size_t)8 * 64 * HW_ * 4);     // 32 MB
    float*    msk   = (float*)alloc((size_t)8 * HW_ * 4);
    _Float16* wpm   = (_Float16*)alloc((size_t)2304 * 64 * 2);
    _Float16* wpih  = (_Float16*)alloc((size_t)576 * 64 * 2);
    _Float16* wpih1 = (_Float16*)alloc((size_t)64 * 64 * 2);
    _Float16* wphh  = (_Float16*)alloc((size_t)576 * 32 * 2);
    _Float16* wphh1 = (_Float16*)alloc((size_t)64 * 32 * 2);
    float*    sf    = (float*)alloc(256);
    float*    ab    = (float*)alloc(6 * 128 * 4);  // 6 slots of [alpha64|beta64]
    _Float16* gact  = (_Float16*)alloc((size_t)HW_ * 64 * 2);
    float*    gtmp  = (float*)alloc((size_t)HW_ * 64 * 4);
    float*    rz    = (float*)alloc((size_t)HW_ * 64 * 4);
    float*    cb    = (float*)alloc((size_t)HW_ * 32 * 4);
    float*    hbuf  = (float*)alloc((size_t)HW_ * 32 * 4);

    // ---- mask + initial hidden state ----
    prep_msk_kernel<<<(8 * HW_ + 255) / 256, 256, 0, stream>>>(incm, msk);
    copyf_kernel<<<(32 * HW_ + 255) / 256, 256, 0, stream>>>(r0, hbuf, 32 * HW_);

    // ---- main 3x3 branch: xn = prelu(bn(conv3(act)))*msk ----
    pack_weights<<<64, 256, 0, stream>>>(c3_w, wpm, sf, 256, 9);
    make_branch_affine<<<1, 64, 0, stream>>>(sf, c3_bng, c3_bnb, c3_bnm, c3_bnv, c3_lb2,
                                             ab + 0, ab + 64, 64);
    prep_act_main<<<131072, 256, 0, stream>>>(x, f, s, c3_lb1, act);
    conv_wmma<256, 3, 3, 1, 4><<<2048, 128, 0, stream>>>(act, wpm, ab + 0, ab + 64, c3_pr,
                                                         msk, nullptr, xn, 131072, 0);

    // ---- main 1x1 branch: xn += prelu(bn(conv1(act))) ----
    pack_weights<<<64, 256, 0, stream>>>(c1_w, wpm, sf, 256, 1);
    make_branch_affine<<<1, 64, 0, stream>>>(sf, c1_bng, c1_bnb, c1_bnm, c1_bnv, c1_lb2,
                                             ab + 128, ab + 192, 64);
    prep_act_main<<<131072, 256, 0, stream>>>(x, f, s, c1_lb1, act);
    conv_wmma<256, 1, 1, 0, 4><<<2048, 128, 0, stream>>>(act, wpm, ab + 128, ab + 192, c1_pr,
                                                         nullptr, nullptr, xn, 131072, 1);

    // 'a' half of the output
    copy_a_kernel<<<(8 * 32 * HW_ + 255) / 256, 256, 0, stream>>>(xn, out);

    // ---- GRU weight prep (once) ----
    pack_weights<<<64, 256, 0, stream>>>(ih_w, wpih, sf, 64, 9);
    make_gate_affine<<<1, 64, 0, stream>>>(sf, ih_b, ih_lb2, ab + 256, ab + 320, 64);
    pack_weights<<<64, 256, 0, stream>>>(ih1_w, wpih1, sf, 64, 1);
    make_gate_affine<<<1, 64, 0, stream>>>(sf, ih1_b, ih1_lb2, ab + 384, ab + 448, 64);
    pack_weights<<<32, 256, 0, stream>>>(hh_w, wphh, sf, 64, 9);
    make_gate_affine<<<1, 64, 0, stream>>>(sf, hh_b, hh_lb2, ab + 512, ab + 576, 32);
    pack_weights<<<32, 256, 0, stream>>>(hh1_w, wphh1, sf, 64, 1);
    make_gate_affine<<<1, 64, 0, stream>>>(sf, hh1_b, hh1_lb2, ab + 640, ab + 704, 32);

    // ---- GRU scan over T=8 ----
    for (int t = 0; t < 8; ++t) {
        const float* xn_t  = xn + (size_t)t * 64 * HW_;
        const float* msk_t = msk + (size_t)t * HW_;
        float*       out_t = out + (size_t)t * 64 * HW_;

        // gates r,z : sigmoid(conv3)*m + sigmoid(conv1)
        prep_xh_act<<<HW_, 64, 0, stream>>>(xn_t, hbuf, ih_lb1, gact);
        conv_wmma<64, 3, 3, 1, 4><<<256, 128, 0, stream>>>(gact, wpih, ab + 256, ab + 320,
                                                           nullptr, nullptr, nullptr, gtmp,
                                                           HW_, 2);
        prep_xh_act<<<HW_, 64, 0, stream>>>(xn_t, hbuf, ih1_lb1, gact);
        conv_wmma<64, 1, 1, 0, 4><<<256, 128, 0, stream>>>(gact, wpih1, ab + 384, ab + 448,
                                                           nullptr, msk_t, gtmp, rz,
                                                           HW_, 3);
        // candidate c : tanh(conv3)*m + tanh(conv1)
        prep_xr_act<<<HW_, 64, 0, stream>>>(xn_t, hbuf, rz, hh_lb1, gact);
        conv_wmma<64, 3, 3, 1, 2><<<256, 128, 0, stream>>>(gact, wphh, ab + 512, ab + 576,
                                                           nullptr, nullptr, nullptr, gtmp,
                                                           HW_, 4);
        prep_xr_act<<<HW_, 64, 0, stream>>>(xn_t, hbuf, rz, hh1_lb1, gact);
        conv_wmma<64, 1, 1, 0, 2><<<256, 128, 0, stream>>>(gact, wphh1, ab + 640, ab + 704,
                                                           nullptr, msk_t, gtmp, cb,
                                                           HW_, 5);
        // h = (1-z)*h + z*c ; write output channels 32..63 for step t
        gru_update<<<(32 * HW_ + 255) / 256, 256, 0, stream>>>(hbuf, rz, cb, out_t);
    }

    // r_out = final hidden state
    copyf_kernel<<<(32 * HW_ + 255) / 256, 256, 0, stream>>>(
        hbuf, out + (size_t)8 * 64 * HW_, 32 * HW_);
}
